// AttentionModel_84327387889814
// MI455X (gfx1250) — matrix operand order repaired and checked by reading, not compiled, not run on hardware
//
#include <hip/hip_runtime.h>
#include <hip/hip_bf16.h>

// ---------------------------------------------------------------------------
// CDNA5 (gfx1250) attention-LSTM captioner.
// All GEMMs: C = A(fp32) x B(fp32)^T (+bias), double-buffered async
// global->LDS staging (ASYNCcnt) feeding v_wmma_f32_16x16x32_bf16.
// ---------------------------------------------------------------------------

typedef __attribute__((ext_vector_type(16))) __bf16 v16bf;
typedef __attribute__((ext_vector_type(8)))  float  v8f;
typedef int b128_t __attribute__((vector_size(16)));   // matches builtin param type

#define AS1 __attribute__((address_space(1)))
#define AS3 __attribute__((address_space(3)))

#if defined(__has_builtin)
#  if __has_builtin(__builtin_amdgcn_global_load_async_to_lds_b128)
#    define HAVE_ASYNC 1
#  endif
#endif
#ifndef HAVE_ASYNC
#  define HAVE_ASYNC 0
#endif

#if HAVE_ASYNC
#  if __has_builtin(__builtin_amdgcn_s_wait_asynccnt)
#    define WAIT_ASYNC(n) __builtin_amdgcn_s_wait_asynccnt(n)
#  else
#    define WAIT_ASYNC(n) asm volatile("s_wait_asynccnt %0" :: "i"(n) : "memory")
#  endif
#else
#  define WAIT_ASYNC(n) ((void)0)
#endif

// One 16-byte chunk: global fp32 -> LDS (async on CDNA5, sync fallback).
__device__ __forceinline__ void stage_b128(const float* g, float* l)
{
#if HAVE_ASYNC
    __builtin_amdgcn_global_load_async_to_lds_b128((AS1 b128_t*)g, (AS3 b128_t*)l,
                                                   /*offset=*/0, /*cpol=*/0);
#else
    *(float4*)l = *(const float4*)g;
#endif
}

// Pack 16 fp32 into a bf16 WMMA fragment (lowers to v_cvt_pk_bf16_f32 pairs,
// which co-execute with XDL WMMA).
__device__ __forceinline__ v16bf pack16(const float4& a, const float4& b,
                                        const float4& c, const float4& d)
{
    v16bf r;
    r[0]  = (__bf16)a.x; r[1]  = (__bf16)a.y; r[2]  = (__bf16)a.z; r[3]  = (__bf16)a.w;
    r[4]  = (__bf16)b.x; r[5]  = (__bf16)b.y; r[6]  = (__bf16)b.z; r[7]  = (__bf16)b.w;
    r[8]  = (__bf16)c.x; r[9]  = (__bf16)c.y; r[10] = (__bf16)c.z; r[11] = (__bf16)c.w;
    r[12] = (__bf16)d.x; r[13] = (__bf16)d.y; r[14] = (__bf16)d.z; r[15] = (__bf16)d.w;
    return r;
}

#define BM 128
#define BN 64
#define BK 32
#define PITCH 36   // fp32/row in LDS (144B) -> spreads the 64x4B banks, keeps 16B align

// C[M,N] = concat_A[M,K] * concat_B[N,K]^T + bias
//   A cols k<K1 from A1(lda1), k>=K1 from A2(lda2) (offset k-K1); same for B1/B2.
//   bias: optional vector biasV[N]; optional matrix biasM[(m/biasRowDiv), N].
// Call-site invariants: K % BK == 0, K1 % BK == 0 (uniform per-tile source select).
__global__ __launch_bounds__(256) void gemm_bf16_wmma(
    const float* __restrict__ A1, int lda1,
    const float* __restrict__ A2, int lda2,
    const float* __restrict__ B1, int ldb1,
    const float* __restrict__ B2, int ldb2,
    int K1,
    const float* __restrict__ biasV,
    const float* __restrict__ biasM, int ldbm, int biasRowDiv,
    float* __restrict__ C, int ldc,
    int M, int N, int K)
{
    __shared__ float lA[2][BM][PITCH];   // 36,864 B
    __shared__ float lB[2][BN][PITCH];   // 18,432 B   (total 54 KB)

    const int tid  = threadIdx.x;
    const int lane = tid & 31;
    const int wave = tid >> 5;
    const int wr   = (wave >> 1) * 32;   // 8 waves -> 4x2 grid of 32x32 quadrants
    const int wc   = (wave & 1)  * 32;
    const int half = lane >> 4;
    const int lrow = lane & 15;

    const int m0 = blockIdx.y * BM;
    const int n0 = blockIdx.x * BN;
    const int nk = K / BK;

    // ---- hoist per-thread staging addresses (both sources) out of the loop ----
    // Thread covers: A chunks i=0..3 (128 rows x 8 float4), B chunks i=0..1.
    const float* pA1[4]; const float* pA2[4]; float* plA[4];
    const float* pB1[2]; const float* pB2[2]; float* plB[2];
    const float* A2s = A2 ? A2 : A1;         // never dereferenced when unused
    const float* B2s = B2 ? B2 : B1;
#pragma unroll
    for (int i = 0; i < 4; ++i) {
        int v  = tid + (i << 8);
        int r  = v >> 3;
        int kc = (v & 7) << 2;
        int gm = m0 + r; if (gm > M - 1) gm = M - 1;   // clamp, keep EXEC full
        pA1[i] = A1  + (size_t)gm * lda1 + kc;
        pA2[i] = A2s + (size_t)gm * lda2 + kc;
        plA[i] = &lA[0][r][kc];
    }
#pragma unroll
    for (int i = 0; i < 2; ++i) {
        int v  = tid + (i << 8);
        int r  = v >> 3;
        int kc = (v & 7) << 2;
        int gn = n0 + r; if (gn > N - 1) gn = N - 1;   // clamp; epilogue guards
        pB1[i] = B1  + (size_t)gn * ldb1 + kc;
        pB2[i] = B2s + (size_t)gn * ldb2 + kc;
        plB[i] = &lB[0][r][kc];
    }
    const size_t ldsBufA = (size_t)BM * PITCH;   // fp32 elements per A buffer
    const size_t ldsBufB = (size_t)BN * PITCH;

    v8f acc[2][2];
#pragma unroll
    for (int i = 0; i < 2; ++i)
#pragma unroll
        for (int j = 0; j < 2; ++j)
            acc[i][j] = (v8f){0.f,0.f,0.f,0.f,0.f,0.f,0.f,0.f};

    // Stage tile kt into LDS buffer `buf`: 6 staging instructions per thread
    // -> per-wave ASYNCcnt += 6 (EXEC always full, so the count is exact).
    auto stage = [&](int kt, int buf) {
        const int  k0 = kt * BK;
        const bool s1 = k0 < K1;                 // uniform per tile (K1 % BK == 0)
        const int  kk = s1 ? k0 : k0 - K1;
        const size_t lb = buf ? ldsBufA : 0;
        const size_t lbB = buf ? ldsBufB : 0;
#pragma unroll
        for (int i = 0; i < 4; ++i)
            stage_b128((s1 ? pA1[i] : pA2[i]) + kk, plA[i] + lb);
#pragma unroll
        for (int i = 0; i < 2; ++i)
            stage_b128((s1 ? pB1[i] : pB2[i]) + kk, plB[i] + lbB);
    };

    stage(0, 0);
    for (int kt = 0; kt < nk; ++kt) {
        const int cur = kt & 1;
        if (kt + 1 < nk) {
            stage(kt + 1, cur ^ 1);   // prefetch next tile into other buffer
            WAIT_ASYNC(6);            // newest 6 may be in flight; tile kt is done
        } else {
            WAIT_ASYNC(0);
        }
        __syncthreads();              // all waves' copies for tile kt resident

        // ---- load all fragment data first (batch the ds_loads), then convert ----
        // A 16x32 frag: lane<16 holds K{0-7}(V0-3)+K{16-23}(V4-7); lane>=16: +8
        // B 32x16 frag: lane<16 holds K{0-7}+K{16-23} of its column; lane>=16: +8
        float4 ar[2][4], br[2][4];
#pragma unroll
        for (int mi = 0; mi < 2; ++mi) {
            const float* pa = &lA[cur][wr + mi * 16 + lrow][half * 8];
            ar[mi][0] = *(const float4*)pa;
            ar[mi][1] = *(const float4*)(pa + 4);
            ar[mi][2] = *(const float4*)(pa + 16);
            ar[mi][3] = *(const float4*)(pa + 20);
        }
#pragma unroll
        for (int ni = 0; ni < 2; ++ni) {
            const float* pb = &lB[cur][wc + ni * 16 + lrow][half * 8];
            br[ni][0] = *(const float4*)pb;
            br[ni][1] = *(const float4*)(pb + 4);
            br[ni][2] = *(const float4*)(pb + 16);
            br[ni][3] = *(const float4*)(pb + 20);
        }
        v16bf af[2], bfr[2];
#pragma unroll
        for (int mi = 0; mi < 2; ++mi)
            af[mi]  = pack16(ar[mi][0], ar[mi][1], ar[mi][2], ar[mi][3]);
#pragma unroll
        for (int ni = 0; ni < 2; ++ni)
            bfr[ni] = pack16(br[ni][0], br[ni][1], br[ni][2], br[ni][3]);

#pragma unroll
        for (int mi = 0; mi < 2; ++mi)
#pragma unroll
            for (int ni = 0; ni < 2; ++ni)
                acc[mi][ni] = __builtin_amdgcn_wmma_f32_16x16x32_bf16(
                    false, af[mi], false, bfr[ni], (short)0, acc[mi][ni],
                    false, false);
        __syncthreads();              // safe to overwrite buffer cur next iter
    }

    // ---- epilogue: lane L, VGPR e -> row = e + 8*(L>=16), col = L%16 ----
#pragma unroll
    for (int mi = 0; mi < 2; ++mi) {
#pragma unroll
        for (int ni = 0; ni < 2; ++ni) {
            int col = n0 + wc + ni * 16 + lrow;
            if (col >= N) continue;
            float bv = biasV ? biasV[col] : 0.f;
            int rbase = m0 + wr + mi * 16 + half * 8;
#pragma unroll
            for (int e = 0; e < 8; ++e) {
                int row = rbase + e;
                if (row < M) {
                    float x = acc[mi][ni][e] + bv;
                    if (biasM)
                        x += biasM[(size_t)(row / biasRowDiv) * ldbm + col];
                    C[(size_t)row * ldc + col] = x;
                }
            }
        }
    }
}

// ---------------------------------------------------------------------------
// Elementwise LSTM cell (torch gate order i,f,g,o).
// ---------------------------------------------------------------------------
__global__ void lstm_cell_kernel(const float* __restrict__ gates,
                                 float* __restrict__ cbuf,
                                 float* __restrict__ hbuf,
                                 float* __restrict__ hiddens,
                                 int t)
{
    int idx = blockIdx.x * blockDim.x + threadIdx.x;  // 0 .. 256*512-1
    if (idx >= 256 * 512) return;
    int b = idx >> 9;
    int j = idx & 511;
    const float* g = gates + (size_t)b * 2048;
    float gi = g[j], gf = g[j + 512], gg = g[j + 1024], go = g[j + 1536];
    float si = 1.f / (1.f + __expf(-gi));
    float sf = 1.f / (1.f + __expf(-gf));
    float so = 1.f / (1.f + __expf(-go));
    float cn = sf * cbuf[idx] + si * tanhf(gg);
    float h  = so * tanhf(cn);
    cbuf[idx] = cn;
    hbuf[idx] = h;
    hiddens[((size_t)b * 128 + t) * 512 + j] = h;
}

// bsum = b_ih + b_hh ; zero-init h0/c0 (ws is poisoned, never restored)
__global__ void prep_kernel(const float* __restrict__ b_ih,
                            const float* __restrict__ b_hh,
                            float* __restrict__ bsum,
                            float* __restrict__ hbuf,
                            float* __restrict__ cbuf)
{
    int i = blockIdx.x * blockDim.x + threadIdx.x;
    if (i < 2048) bsum[i] = b_ih[i] + b_hh[i];
    if (i < 256 * 512) { hbuf[i] = 0.f; cbuf[i] = 0.f; }
}

extern "C" void kernel_launch(void* const* d_in, const int* in_sizes, int n_in,
                              void* d_out, int out_size, void* d_ws, size_t ws_size,
                              hipStream_t stream)
{
    const float* images   = (const float*)d_in[0];   // [256, 2048]
    const float* captions = (const float*)d_in[1];   // [256, 128, 512]
    const float* W_fc     = (const float*)d_in[2];   // [256, 2048]
    const float* b_fc     = (const float*)d_in[3];   // [256]
    const float* W_att    = (const float*)d_in[4];   // [256, 768]
    const float* b_att    = (const float*)d_in[5];   // [256]
    const float* W_ih     = (const float*)d_in[6];   // [2048, 512]
    const float* b_ih     = (const float*)d_in[7];   // [2048]
    const float* W_hh     = (const float*)d_in[8];   // [2048, 512]
    const float* b_hh     = (const float*)d_in[9];   // [2048]
    const float* W_out    = (const float*)d_in[10];  // [1004, 512]
    const float* b_out    = (const float*)d_in[11];  // [1004]
    float* out = (float*)d_out;                      // [256, 128, 1004]

    // workspace layout (fp32 elements), ~26.6M floats (~106 MB)
    float* ws      = (float*)d_ws;
    float* feats   = ws;                   // 256*256
    float* Fatt    = feats   + 65536;      // 256*256
    float* ctx     = Fatt    + 65536;      // 256*128*256
    float* Fih     = ctx     + 8388608;    // 256*2048
    float* gates   = Fih     + 524288;     // 256*2048
    float* bsum    = gates   + 524288;     // 2048
    float* cbuf    = bsum    + 2048;       // 256*512
    float* hbuf    = cbuf    + 131072;     // 256*512
    float* hiddens = hbuf    + 131072;     // 256*128*512

    const dim3 blk(256);

    // 1) feats = images @ W_fc^T + b_fc            (M=256, N=256, K=2048)
    gemm_bf16_wmma<<<dim3(4, 2), blk, 0, stream>>>(
        images, 2048, nullptr, 0, W_fc, 2048, nullptr, 0, /*K1=*/2048,
        b_fc, nullptr, 0, 1, feats, 256, 256, 256, 2048);

    // 2) Fatt = feats @ W_att[:, :256]^T + b_att   (M=256, N=256, K=256)
    gemm_bf16_wmma<<<dim3(4, 2), blk, 0, stream>>>(
        feats, 256, nullptr, 0, W_att, 768, nullptr, 0, /*K1=*/256,
        b_att, nullptr, 0, 1, Fatt, 256, 256, 256, 256);

    // 3) ctx[b,t,:] = captions @ W_att[:, 256:]^T + Fatt[b]  (M=32768, N=256, K=512)
    gemm_bf16_wmma<<<dim3(4, 256), blk, 0, stream>>>(
        captions, 512, nullptr, 0, W_att + 256, 768, nullptr, 0, /*K1=*/512,
        nullptr, Fatt, 256, /*biasRowDiv=*/128, ctx, 256, 32768, 256, 512);

    // 4) bsum = b_ih + b_hh ; h0 = c0 = 0
    prep_kernel<<<512, 256, 0, stream>>>(b_ih, b_hh, bsum, hbuf, cbuf);

    // 5) Fih = feats @ W_ih[:, :256]^T + bsum      (M=256, N=2048, K=256)
    gemm_bf16_wmma<<<dim3(32, 2), blk, 0, stream>>>(
        feats, 256, nullptr, 0, W_ih, 512, nullptr, 0, /*K1=*/256,
        bsum, nullptr, 0, 1, Fih, 2048, 256, 2048, 256);

    // 6) recurrence: gates = [ctx_t | h] @ [W_ih[:,256:] | W_hh]^T + Fih
    for (int t = 0; t < 128; ++t) {
        gemm_bf16_wmma<<<dim3(32, 2), blk, 0, stream>>>(
            ctx + (size_t)t * 256, /*lda1=*/128 * 256,   // ctx[b][t][:], stride T*256
            hbuf, 512,
            W_ih + 256, 512,                             // W_ih columns 256..511
            W_hh, 512,
            /*K1=*/256,
            nullptr, Fih, 2048, /*biasRowDiv=*/1,
            gates, 2048, 256, 2048, 768);
        lstm_cell_kernel<<<512, 256, 0, stream>>>(gates, cbuf, hbuf, hiddens, t);
    }

    // 7) out = hiddens @ W_out^T + b_out           (M=32768, N=1004, K=512)
    gemm_bf16_wmma<<<dim3(16, 256), blk, 0, stream>>>(
        hiddens, 512, nullptr, 0, W_out, 512, nullptr, 0, /*K1=*/512,
        b_out, nullptr, 0, 1, out, 1004, 32768, 1004, 512);
}
